// LocallySelfAttention_85993835200852
// MI455X (gfx1250) — matrix operand order repaired and checked by reading, not compiled, not run on hardware
//
#include <hip/hip_runtime.h>
#include <hip/hip_bf16.h>

// ---------------------------------------------------------------------------
// Locally-windowed multi-head self-attention for gfx1250 (CDNA5, wave32).
// f16 data / f32 accumulate via v_wmma_f32_16x16x32_f16.
// ---------------------------------------------------------------------------

typedef _Float16 v16h __attribute__((ext_vector_type(16)));
typedef _Float16 h8   __attribute__((ext_vector_type(8)));
typedef _Float16 h4   __attribute__((ext_vector_type(4)));
typedef float    v8f  __attribute__((ext_vector_type(8)));

#define SEQ   2048
#define EMB   768
#define NH    12
#define HD    64
#define BATCH 4
#define WIN   256
#define MROWS (BATCH * SEQ)       // 8192
#define LOG2E 1.4426950408889634f

__device__ __forceinline__ v8f wmma_f16(v16h a, v16h b, v8f c) {
  return __builtin_amdgcn_wmma_f32_16x16x32_f16(false, a, false, b, (short)0, c,
                                                false, false);
}

__device__ __forceinline__ v8f zero8() {
  v8f r;
#pragma unroll
  for (int i = 0; i < 8; ++i) r[i] = 0.0f;
  return r;
}

// A-fragment (16x32 f16): elements 0..15 = row[kb..kb+15]
__device__ __forceinline__ v16h frag_a(const _Float16* __restrict__ row, int kb) {
  h8 lo = *(const h8*)(row + kb);
  h8 hi = *(const h8*)(row + kb + 8);
  v16h r;
#pragma unroll
  for (int i = 0; i < 8; ++i) { r[i] = lo[i]; r[i + 8] = hi[i]; }
  return r;
}

// B-fragment (32x16 f16): 16 contiguous halves per lane
__device__ __forceinline__ v16h frag_b(const _Float16* __restrict__ p) {
  h8 lo = *(const h8*)(p);
  h8 hi = *(const h8*)(p + 8);
  v16h r;
#pragma unroll
  for (int i = 0; i < 8; ++i) { r[i] = lo[i]; r[i + 8] = hi[i]; }
  return r;
}

// ---------------------------------------------------------------------------
// f32 -> f16 conversion (128b load, 64b packed store)
// ---------------------------------------------------------------------------
__global__ __launch_bounds__(256) void cvt_f32_f16(const float* __restrict__ in,
                                                   _Float16* __restrict__ out,
                                                   int n4) {
  int i = blockIdx.x * blockDim.x + threadIdx.x;
  if (i >= n4) return;
  float4 v = ((const float4*)in)[i];
  h4 o = { (_Float16)v.x, (_Float16)v.y, (_Float16)v.z, (_Float16)v.w };
  *(h4*)(out + (size_t)i * 4) = o;
}

// ---------------------------------------------------------------------------
// GEMM: C[M=8192, N=768] = X[M,768] @ W^T + bias ; W is [N,768] row-major.
// Each wave computes a 32x64 tile (2 A-frags, 8 accumulators, 8 WMMA/k-step).
// mode 0: write f16 [B,H,S,D]   (Q, K)
// mode 1: write f16 [B,H,D,S]   (V transposed for P·V B-fragments)
// mode 2: write f32 row-major   (final output)
// ---------------------------------------------------------------------------
__global__ __launch_bounds__(256) void gemm_wmma(
    const _Float16* __restrict__ X, const _Float16* __restrict__ W,
    const float* __restrict__ bias, _Float16* __restrict__ out16,
    float* __restrict__ out32, int mode) {
  const int lane = threadIdx.x & 31;
  const int wave = threadIdx.x >> 5;
  const int gw = blockIdx.x * 8 + wave;         // 3072 waves total
  const int m0 = (gw / 12) * 32;
  const int n0 = (gw % 12) * 64;
  const int half = lane >> 4;
  const int c = lane & 15;
  const int kbA = half * 16;

  const _Float16* xrow0 = X + (size_t)(m0 + c) * EMB;        // A rows, group 0
  const _Float16* xrow1 = X + (size_t)(m0 + 16 + c) * EMB;   // A rows, group 1

  const _Float16* wp[4];
  float bv[4];
#pragma unroll
  for (int t = 0; t < 4; ++t) {
    int n = n0 + t * 16 + c;
    wp[t] = W + (size_t)n * EMB + half * 16;
    bv[t] = bias[n];
  }

  v8f acc0[4], acc1[4];
#pragma unroll
  for (int t = 0; t < 4; ++t) { acc0[t] = zero8(); acc1[t] = zero8(); }

#pragma unroll 2
  for (int k0 = 0; k0 < EMB; k0 += 32) {
    v16h a0 = frag_a(xrow0 + k0, kbA);
    v16h a1 = frag_a(xrow1 + k0, kbA);
#pragma unroll
    for (int t = 0; t < 4; ++t) {
      v16h b = frag_b(wp[t] + k0);
      acc0[t] = wmma_f16(a0, b, acc0[t]);
      acc1[t] = wmma_f16(a1, b, acc1[t]);
    }
  }

#pragma unroll
  for (int g = 0; g < 2; ++g) {
    const v8f* acc = g ? acc1 : acc0;
#pragma unroll
    for (int r = 0; r < 8; ++r) {
      int mg = m0 + g * 16 + r + half * 8;   // global row
      int bb = mg >> 11;                     // batch
      int ss = mg & (SEQ - 1);               // seq position
#pragma unroll
      for (int t = 0; t < 4; ++t) {
        float val = acc[t][r] + bv[t];
        int n = n0 + t * 16 + c;
        if (mode == 2) {
          out32[(size_t)mg * EMB + n] = val;
        } else {
          int h = n >> 6, d = n & 63;
          size_t idx;
          if (mode == 0)
            idx = (((size_t)bb * NH + h) * SEQ + ss) * HD + d;   // [B,H,S,D]
          else
            idx = (((size_t)bb * NH + h) * HD + d) * SEQ + ss;   // [B,H,D,S]
          out16[idx] = (_Float16)val;
        }
      }
    }
  }
}

// ---------------------------------------------------------------------------
// Windowed flash attention. One wave per (b,h,16-row query tile).
// Key loop over 32-key steps spanning [q0-WIN, q0+15+WIN].
// ---------------------------------------------------------------------------
#define PSTR 40  // LDS row stride in halves (80B: 16B-aligned rows, conflict-light)

__global__ __launch_bounds__(256) void attn_wmma(
    const _Float16* __restrict__ Q, const _Float16* __restrict__ K,
    const _Float16* __restrict__ Vt, _Float16* __restrict__ Aout) {
  __shared__ _Float16 Pl[8][16 * PSTR];

  const int lane = threadIdx.x & 31;
  const int wave = threadIdx.x >> 5;
  const int gw = blockIdx.x * 8 + wave;     // 6144 waves total
  const int q0 = (gw & 127) << 4;           // query tile start
  const int bh = gw >> 7;                   // b*NH + h
  const int half = lane >> 4;
  const int c = lane & 15;

  const _Float16* Qb = Q + (size_t)bh * SEQ * HD;
  const _Float16* Kb = K + (size_t)bh * SEQ * HD;
  const _Float16* Vb = Vt + (size_t)bh * HD * SEQ;
  _Float16* Pw = &Pl[wave][0];

  // Preload Q A-fragments for d=[0,32) and d=[32,64)
  const _Float16* qrow = Qb + (size_t)(q0 + c) * HD;
  const v16h Aq0 = frag_a(qrow, half * 16);
  const v16h Aq1 = frag_a(qrow + 32, half * 16);

  float mrow[8], lrow[8];
  v8f O[4];
#pragma unroll
  for (int r = 0; r < 8; ++r) { mrow[r] = -3.0e38f; lrow[r] = 0.0f; }
#pragma unroll
  for (int t = 0; t < 4; ++t) O[t] = zero8();

  int kstart = q0 - WIN; if (kstart < 0) kstart = 0;
  kstart &= ~31;
  int kend = q0 + 16 + WIN; if (kend > SEQ) kend = SEQ;

  for (int j0 = kstart; j0 < kend; j0 += 32) {
    // ---- scores: S = (Q/tau) K^T, two 16x16 key sub-tiles ----
    const int na = j0 + c, nb = j0 + 16 + c;
    const int nac = na < SEQ ? na : SEQ - 1;
    const int nbc = nb < SEQ ? nb : SEQ - 1;
    const _Float16* kpa = Kb + (size_t)nac * HD + half * 16;
    const _Float16* kpb = Kb + (size_t)nbc * HD + half * 16;

    // Prefetch next key tile (predictable stride; speculative is safe)
    __builtin_prefetch(kpa + 32 * HD, 0, 1);
    __builtin_prefetch(kpb + 32 * HD, 0, 1);

    v8f s0 = zero8(), s1 = zero8();
    s0 = wmma_f16(Aq0, frag_b(kpa), s0);
    s0 = wmma_f16(Aq1, frag_b(kpa + 32), s0);
    s1 = wmma_f16(Aq0, frag_b(kpb), s1);
    s1 = wmma_f16(Aq1, frag_b(kpb + 32), s1);

    __builtin_amdgcn_wave_barrier();  // WAR fence vs. previous iter's P loads

    // ---- online softmax (rows r+8*half live across the 16-lane half-group) ----
#pragma unroll
    for (int r = 0; r < 8; ++r) {
      const int row = q0 + r + half * 8;
      const bool va = (na < SEQ) && (row - na <= WIN) && (na - row <= WIN);
      const bool vb = (nb < SEQ) && (row - nb <= WIN) && (nb - row <= WIN);
      float fa = va ? s0[r] * 0.125f : -3.0e38f;   // tau = sqrt(64) = 8
      float fb = vb ? s1[r] * 0.125f : -3.0e38f;
      float mx = fmaxf(fa, fb);
      mx = fmaxf(mx, __shfl_xor(mx, 1, 32));
      mx = fmaxf(mx, __shfl_xor(mx, 2, 32));
      mx = fmaxf(mx, __shfl_xor(mx, 4, 32));
      mx = fmaxf(mx, __shfl_xor(mx, 8, 32));
      const float mnew = fmaxf(mrow[r], mx);
      const float alpha = exp2f((mrow[r] - mnew) * LOG2E);
      const float pa = va ? exp2f((fa - mnew) * LOG2E) : 0.0f;
      const float pb = vb ? exp2f((fb - mnew) * LOG2E) : 0.0f;
      float ps = pa + pb;
      ps += __shfl_xor(ps, 1, 32);
      ps += __shfl_xor(ps, 2, 32);
      ps += __shfl_xor(ps, 4, 32);
      ps += __shfl_xor(ps, 8, 32);
      lrow[r] = lrow[r] * alpha + ps;
      mrow[r] = mnew;
#pragma unroll
      for (int t = 0; t < 4; ++t) O[t][r] *= alpha;
      const int mr = r + half * 8;
      Pw[mr * PSTR + c] = (_Float16)pa;        // C-layout -> row-major staging
      Pw[mr * PSTR + c + 16] = (_Float16)pb;
    }

    __builtin_amdgcn_wave_barrier();  // RAW fence: stores above -> loads below

    // ---- O += P V : P as A-fragment (k = 32 keys), Vt gives contiguous B ----
    const v16h Ap = frag_a(Pw + c * PSTR, half * 16);
    int js = j0 + half * 16;
    if (js > SEQ - 16) js = SEQ - 16;  // clamp (masked cols have p == 0)
#pragma unroll
    for (int t = 0; t < 4; ++t) {
      const _Float16* vp = Vb + (size_t)(t * 16 + c) * SEQ + js;
      __builtin_prefetch(vp + 32, 0, 1);       // next key step for this V row
      O[t] = wmma_f16(Ap, frag_b(vp), O[t]);
    }
  }

  // ---- epilogue: normalize, write f16 [B,S,E] for the output projection ----
  const int b = bh / NH, h = bh % NH;
#pragma unroll
  for (int r = 0; r < 8; ++r) {
    const float inv = lrow[r] > 0.0f ? 1.0f / lrow[r] : 0.0f;
    const int row = q0 + r + half * 8;
    _Float16* op = Aout + ((size_t)b * SEQ + row) * EMB + h * HD + c;
#pragma unroll
    for (int t = 0; t < 4; ++t) op[t * 16] = (_Float16)(O[t][r] * inv);
  }
}

// ---------------------------------------------------------------------------
// Host-side launch
// ---------------------------------------------------------------------------
extern "C" void kernel_launch(void* const* d_in, const int* in_sizes, int n_in,
                              void* d_out, int out_size, void* d_ws, size_t ws_size,
                              hipStream_t stream) {
  (void)in_sizes; (void)n_in; (void)out_size; (void)ws_size;

  const float* x  = (const float*)d_in[0];
  const float* Wq = (const float*)d_in[1];
  const float* bq = (const float*)d_in[2];
  const float* Wk = (const float*)d_in[3];
  const float* bk = (const float*)d_in[4];
  const float* Wv = (const float*)d_in[5];
  const float* bv = (const float*)d_in[6];
  const float* Wo = (const float*)d_in[7];
  const float* bo = (const float*)d_in[8];
  float* out = (float*)d_out;

  const size_t NX = (size_t)MROWS * EMB;   // 6,291,456
  const size_t NW = (size_t)EMB * EMB;     //   589,824

  char* w = (char*)d_ws;
  _Float16* xh   = (_Float16*)w; w += NX * 2;
  _Float16* Wq16 = (_Float16*)w; w += NW * 2;
  _Float16* Wk16 = (_Float16*)w; w += NW * 2;
  _Float16* Wv16 = (_Float16*)w; w += NW * 2;
  _Float16* Wo16 = (_Float16*)w; w += NW * 2;
  _Float16* Qh   = (_Float16*)w; w += NX * 2;
  _Float16* Kh   = (_Float16*)w; w += NX * 2;
  _Float16* Vt   = (_Float16*)w; w += NX * 2;
  _Float16* At   = (_Float16*)w; w += NX * 2;

  // f32 -> f16 conversions
  {
    int n4 = (int)(NX / 4);
    cvt_f32_f16<<<(n4 + 255) / 256, 256, 0, stream>>>(x, xh, n4);
    int w4 = (int)(NW / 4);
    cvt_f32_f16<<<(w4 + 255) / 256, 256, 0, stream>>>(Wq, Wq16, w4);
    cvt_f32_f16<<<(w4 + 255) / 256, 256, 0, stream>>>(Wk, Wk16, w4);
    cvt_f32_f16<<<(w4 + 255) / 256, 256, 0, stream>>>(Wv, Wv16, w4);
    cvt_f32_f16<<<(w4 + 255) / 256, 256, 0, stream>>>(Wo, Wo16, w4);
  }

  // Projections: 3072 waves (32-row tiles) = 384 blocks x 8 waves
  const int GB_GEMM = (MROWS / 32) * (EMB / 64) / 8;  // 384
  gemm_wmma<<<GB_GEMM, 256, 0, stream>>>(xh, Wq16, bq, Qh, nullptr, 0);
  gemm_wmma<<<GB_GEMM, 256, 0, stream>>>(xh, Wk16, bk, Kh, nullptr, 0);
  gemm_wmma<<<GB_GEMM, 256, 0, stream>>>(xh, Wv16, bv, Vt, nullptr, 1);

  // Attention: 4*12*128 = 6144 query tiles = 768 blocks x 8 waves
  const int GB_ATTN = (BATCH * NH * (SEQ / 16)) / 8;  // 768
  attn_wmma<<<GB_ATTN, 256, 0, stream>>>(Qh, Kh, Vt, At);

  // Output projection (f32 out + bias)
  gemm_wmma<<<GB_GEMM, 256, 0, stream>>>(At, Wo16, bo, nullptr, out, 2);
}
